// TiSASRec_5351529250958
// MI455X (gfx1250) — compile-verified, matrix operand order checked by reading, not run on hardware
//
#include <hip/hip_runtime.h>
#include <hip/hip_bf16.h>

// Problem constants (from reference)
#define BB   8
#define LL   200
#define HH_  128
#define NHH  4
#define HDD  32
#define NBB  2
#define MM   (BB * LL)      // 1600 rows
#define TS1  257            // TIME_SPAN + 1
#define QT_LD 272           // pitch for Qt (multiple of 16)
#define PADV (-4294967295.0f)

typedef __bf16 bf16_t;
typedef __attribute__((ext_vector_type(16))) __bf16 v16bf;
typedef __attribute__((ext_vector_type(8)))  __bf16 v8bf;
typedef __attribute__((ext_vector_type(8)))  float  v8f;

// ---------------------------------------------------------------------------
// WMMA GEMM:  C[m,n] = sum_k A[m,k] * W[n,k]  (+bias[n]) (+addsrc[m,n]) (relu?)
// A: bf16 [.., lda]; W: bf16 [N rows, ldw] (row-major along k) -> B = W^T.
// One wave computes a 16 x (16*NT) tile: the A fragment is loaded once per
// k-step and reused across NT WMMAs (A is the streamed operand; W is small
// and L2-resident). K must be a multiple of 32; M tiles exact (M=1600).
// ---------------------------------------------------------------------------
template <int NT>
__global__ void __launch_bounds__(32)
wmma_gemm(const bf16_t* __restrict__ A, int lda,
          const bf16_t* __restrict__ W, int ldw,
          const float* __restrict__ bias,
          const float* __restrict__ addsrc, int ldadd,
          float* __restrict__ C, int ldc, bf16_t* __restrict__ Cbf,
          int N, int K, int relu)
{
    const int mt   = blockIdx.x * 16;
    const int nt0  = blockIdx.y * (16 * NT);
    const int lane = threadIdx.x;
    const int half = lane >> 4;     // 0: K 0-7,16-23   1: K 8-15,24-31
    const int l15  = lane & 15;

    int nrow[NT];
#pragma unroll
    for (int j = 0; j < NT; ++j) {
        int r = nt0 + 16 * j + l15;
        nrow[j] = (r > N - 1) ? (N - 1) : r;   // clamp W row loads for ragged N
    }

    v8f acc[NT];
#pragma unroll
    for (int j = 0; j < NT; ++j) acc[j] = (v8f){};

    for (int k0 = 0; k0 < K; k0 += 32) {
        const bf16_t* arow = A + (size_t)(mt + l15) * lda + k0 + half * 8;
        v8bf a1 = *(const v8bf*)(arow);
        v8bf a2 = *(const v8bf*)(arow + 16);
        v16bf af;
#pragma unroll
        for (int i = 0; i < 8; ++i) { af[i] = a1[i]; af[8 + i] = a2[i]; }
#pragma unroll
        for (int j = 0; j < NT; ++j) {
            const bf16_t* wrow = W + (size_t)nrow[j] * ldw + k0 + half * 8;
            v8bf b1 = *(const v8bf*)(wrow);
            v8bf b2 = *(const v8bf*)(wrow + 16);
            v16bf bw;
#pragma unroll
            for (int i = 0; i < 8; ++i) { bw[i] = b1[i]; bw[8 + i] = b2[i]; }
            acc[j] = __builtin_amdgcn_wmma_f32_16x16x32_bf16(
                false, af, false, bw, (short)0, acc[j], false, false);
        }
    }

#pragma unroll
    for (int j = 0; j < NT; ++j) {
        const int nn = nt0 + 16 * j + l15;
        if (nn < N) {
            const float bv = bias ? bias[nn] : 0.0f;
#pragma unroll
            for (int r = 0; r < 8; ++r) {
                const int m = mt + r + half * 8;   // C layout: vgpr r -> row r / r+8
                float v = acc[j][r] + bv;
                if (addsrc) v += addsrc[(size_t)m * ldadd + nn];
                if (relu)   v = fmaxf(v, 0.0f);
                C[(size_t)m * ldc + nn] = v;
                if (Cbf) Cbf[(size_t)m * ldc + nn] = (bf16_t)v;
            }
        }
    }
}

// ---------------------------------------------------------------------------
// LayerNorm over last dim H=128, one wave per row (4 elems/lane).
// ---------------------------------------------------------------------------
__global__ void __launch_bounds__(32)
ln_kernel(const float* __restrict__ x, const float* __restrict__ g,
          const float* __restrict__ beta, float* __restrict__ y,
          bf16_t* __restrict__ ybf)
{
    const int row = blockIdx.x, lane = threadIdx.x;
    const float* xr = x + (size_t)row * HH_;
    float v[4]; float s = 0.0f;
#pragma unroll
    for (int i = 0; i < 4; ++i) { v[i] = xr[lane + 32 * i]; s += v[i]; }
    for (int off = 16; off; off >>= 1) s += __shfl_xor(s, off, 32);
    const float mean = s * (1.0f / HH_);
    float var = 0.0f;
#pragma unroll
    for (int i = 0; i < 4; ++i) { float d = v[i] - mean; var += d * d; }
    for (int off = 16; off; off >>= 1) var += __shfl_xor(var, off, 32);
    var *= (1.0f / HH_);
    const float rstd = rsqrtf(var + 1e-8f);
#pragma unroll
    for (int i = 0; i < 4; ++i) {
        const int h = lane + 32 * i;
        const float o = (v[i] - mean) * rstd * g[h] + beta[h];
        y[(size_t)row * HH_ + h] = o;
        if (ybf) ybf[(size_t)row * HH_ + h] = (bf16_t)o;
    }
}

// ---------------------------------------------------------------------------
// Attention: one wave per (head n, batch b, query q). HD=32 -> one dim per lane.
// Scores: dot(Q, K+posK) via __shfl broadcast + Qt time-table gather; softmax;
// output: sum_k p[k]*(V + posV + timeV[t_k]).
// ---------------------------------------------------------------------------
__global__ void __launch_bounds__(32)
attn_kernel(const float* __restrict__ Q, const float* __restrict__ K,
            const float* __restrict__ V, const float* __restrict__ posK,
            const float* __restrict__ posV, const float* __restrict__ timeV,
            const int* __restrict__ tmat, const int* __restrict__ logs,
            const float* __restrict__ Qt, float* __restrict__ Out)
{
    __shared__ float sm[LL];
    const int n = blockIdx.x, b = blockIdx.y, q = blockIdx.z;
    const int lane = threadIdx.x;
    const int mrow = b * LL + q;
    const bool qpad = (logs[mrow] == 0);
    const float qown = Q[(size_t)mrow * HH_ + n * HDD + lane];
    const float invs = 0.17677669529663687f;   // 1/sqrt(HD)
    const float* QtRow = Qt + ((size_t)n * MM + mrow) * QT_LD;
    const int* tmrow = tmat + (size_t)mrow * LL;

    for (int kb = 0; kb < LL; kb += 32) {
        const int k = kb + lane;
        const bool valid = (k < LL);
        const int kc = valid ? k : 0;
        const float* Krow  = K    + (size_t)(b * LL + kc) * HH_ + n * HDD;
        const float* pKrow = posK + (size_t)kc * HH_ + n * HDD;
        float acc = 0.0f;
#pragma unroll
        for (int d = 0; d < HDD; ++d) {
            const float qd = __shfl(qown, d, 32);   // all lanes active here
            acc += qd * (Krow[d] + pKrow[d]);
        }
        if (valid) {
            float sc;
            if (qpad || k > q) sc = PADV;
            else               sc = (acc + QtRow[tmrow[k]]) * invs;
            sm[k] = sc;
        }
    }
    __syncthreads();

    float mx = -3.4e38f;
    for (int k = lane; k < LL; k += 32) mx = fmaxf(mx, sm[k]);
    for (int off = 16; off; off >>= 1) mx = fmaxf(mx, __shfl_xor(mx, off, 32));
    float sum = 0.0f;
    for (int k = lane; k < LL; k += 32) {
        const float e = __expf(sm[k] - mx);
        sm[k] = e; sum += e;
    }
    for (int off = 16; off; off >>= 1) sum += __shfl_xor(sum, off, 32);
    const float inv = 1.0f / sum;
    __syncthreads();

    const int d = lane;
    float acc = 0.0f;
    for (int k = 0; k < LL; ++k) {
        const float w = sm[k];
        const int t = tmrow[k];
        const float v = V[(size_t)(b * LL + k) * HH_ + n * HDD + d]
                      + posV[(size_t)k * HH_ + n * HDD + d]
                      + timeV[(size_t)t * HH_ + n * HDD + d];
        acc += w * v;
    }
    Out[(size_t)mrow * HH_ + n * HDD + d] = acc * inv;
}

// ---------------------------------------------------------------------------
// Small helper kernels
// ---------------------------------------------------------------------------
__global__ void f2bf(const float* __restrict__ x, bf16_t* __restrict__ y, int n)
{
    const int i = blockIdx.x * blockDim.x + threadIdx.x;
    if (i < n) y[i] = (bf16_t)x[i];
}

__global__ void gather_scale(const int* __restrict__ logs, const float* __restrict__ emb,
                             float* __restrict__ seqs, bf16_t* __restrict__ seqsbf)
{
    const int m = blockIdx.x, h = threadIdx.x;   // blockDim = 128
    const int idx = logs[m];
    const float v = (idx == 0) ? 0.0f
                               : emb[(size_t)idx * HH_ + h] * 11.313708498984761f; // sqrt(H)
    seqs[(size_t)m * HH_ + h] = v;
    seqsbf[(size_t)m * HH_ + h] = (bf16_t)v;
}

__global__ void add_kernel(const float* __restrict__ a, float* __restrict__ b, int n)
{
    const int i = blockIdx.x * blockDim.x + threadIdx.x;
    if (i < n) b[i] = a[i] + b[i];
}

__global__ void maskmul(const float* __restrict__ x, const int* __restrict__ logs,
                        float* __restrict__ y, bf16_t* __restrict__ ybf)
{
    const int m = blockIdx.x, h = threadIdx.x;   // blockDim = 128
    const float v = (logs[m] == 0) ? 0.0f : x[(size_t)m * HH_ + h];
    y[(size_t)m * HH_ + h] = v;
    ybf[(size_t)m * HH_ + h] = (bf16_t)v;
}

__global__ void __launch_bounds__(32)
logits_kernel(const float* __restrict__ feats, const float* __restrict__ emb,
              const int* __restrict__ pos, const int* __restrict__ neg,
              float* __restrict__ out)
{
    const int m = blockIdx.x, lane = threadIdx.x;
    const float* f = feats + (size_t)m * HH_;
    const int pi = pos[m], ni = neg[m];
    float sp = 0.0f, sn = 0.0f;
#pragma unroll
    for (int i = 0; i < 4; ++i) {
        const int h = lane + 32 * i;
        const float fv = f[h];
        sp += fv * emb[(size_t)pi * HH_ + h];
        sn += fv * emb[(size_t)ni * HH_ + h];
    }
    for (int off = 16; off; off >>= 1) {
        sp += __shfl_xor(sp, off, 32);
        sn += __shfl_xor(sn, off, 32);
    }
    if (lane == 0) { out[m] = sp; out[MM + m] = sn; }
}

// ---------------------------------------------------------------------------
// Host orchestration
// ---------------------------------------------------------------------------
extern "C" void kernel_launch(void* const* d_in, const int* in_sizes, int n_in,
                              void* d_out, int out_size, void* d_ws, size_t ws_size,
                              hipStream_t stream)
{
    (void)in_sizes; (void)n_in; (void)out_size; (void)ws_size;

    const int*   log_seqs = (const int*)  d_in[1];
    const int*   tmat     = (const int*)  d_in[2];
    const int*   pos_seqs = (const int*)  d_in[3];
    const int*   neg_seqs = (const int*)  d_in[4];
    const float* item_emb = (const float*)d_in[5];
    const float* posK     = (const float*)d_in[6];
    const float* posV     = (const float*)d_in[7];
    const float* timeK    = (const float*)d_in[8];
    const float* timeV    = (const float*)d_in[9];
    const float* attn_g   = (const float*)d_in[10];
    const float* attn_b   = (const float*)d_in[11];
    const float* Wq       = (const float*)d_in[12];
    const float* bq       = (const float*)d_in[13];
    const float* Wk       = (const float*)d_in[14];
    const float* bk       = (const float*)d_in[15];
    const float* Wv       = (const float*)d_in[16];
    const float* bv       = (const float*)d_in[17];
    const float* fwd_g    = (const float*)d_in[18];
    const float* fwd_b    = (const float*)d_in[19];
    const float* W1       = (const float*)d_in[20];
    const float* b1       = (const float*)d_in[21];
    const float* W2       = (const float*)d_in[22];
    const float* b2       = (const float*)d_in[23];
    const float* last_g   = (const float*)d_in[24];
    const float* last_b   = (const float*)d_in[25];

    char* ws = (char*)d_ws;
    const size_t F = (size_t)MM * HH_ * 4;          // 819200 B per f32 [M,H]
    float*  seqs_f = (float*)(ws + 0 * F);
    float*  qin_f  = (float*)(ws + 1 * F);
    float*  q_f    = (float*)(ws + 2 * F);
    float*  k_f    = (float*)(ws + 3 * F);
    float*  v_f    = (float*)(ws + 4 * F);
    float*  ao_f   = (float*)(ws + 5 * F);
    float*  ff1_f  = (float*)(ws + 6 * F);
    float*  qt_f   = (float*)(ws + 7 * F);          // 4 * 1600 * 272 * 4 B
    size_t  base2  = 7 * F + (size_t)NHH * MM * QT_LD * 4;
    const size_t Bh = (size_t)MM * HH_ * 2;         // 409600 B per bf16 [M,H]
    bf16_t* seqs_b = (bf16_t*)(ws + base2 + 0 * Bh);
    bf16_t* qin_b  = (bf16_t*)(ws + base2 + 1 * Bh);
    bf16_t* q_b    = (bf16_t*)(ws + base2 + 2 * Bh);
    bf16_t* ff1_b  = (bf16_t*)(ws + base2 + 3 * Bh);
    size_t  base3  = base2 + 4 * Bh;
    const int HW = HH_ * HH_;                       // 16384
    const size_t Wb = (size_t)NBB * HW * 2;         // 65536 B per weight set
    bf16_t* wq_b = (bf16_t*)(ws + base3 + 0 * Wb);
    bf16_t* wk_b = (bf16_t*)(ws + base3 + 1 * Wb);
    bf16_t* wv_b = (bf16_t*)(ws + base3 + 2 * Wb);
    bf16_t* w1_b = (bf16_t*)(ws + base3 + 3 * Wb);
    bf16_t* w2_b = (bf16_t*)(ws + base3 + 4 * Wb);
    bf16_t* tk_b = (bf16_t*)(ws + base3 + 5 * Wb);  // 257*128 bf16

    // weight / table conversions to bf16
    const int nW = NBB * HW;
    f2bf<<<(nW + 255) / 256, 256, 0, stream>>>(Wq, wq_b, nW);
    f2bf<<<(nW + 255) / 256, 256, 0, stream>>>(Wk, wk_b, nW);
    f2bf<<<(nW + 255) / 256, 256, 0, stream>>>(Wv, wv_b, nW);
    f2bf<<<(nW + 255) / 256, 256, 0, stream>>>(W1, w1_b, nW);
    f2bf<<<(nW + 255) / 256, 256, 0, stream>>>(W2, w2_b, nW);
    const int nT = TS1 * HH_;
    f2bf<<<(nT + 255) / 256, 256, 0, stream>>>(timeK, tk_b, nT);

    gather_scale<<<MM, HH_, 0, stream>>>(log_seqs, item_emb, seqs_f, seqs_b);

    const dim3 g128(MM / 16, HH_ / 64);            // 100 x 2, NT=4 -> 64 cols/wave
    const dim3 gqt(MM / 16, (TS1 + 63) / 64);      // 100 x 5
    const dim3 gat(NHH, BB, LL);
    const int nEl = MM * HH_;

    for (int i = 0; i < NBB; ++i) {
        ln_kernel<<<MM, 32, 0, stream>>>(seqs_f, attn_g + i * HH_, attn_b + i * HH_,
                                         qin_f, qin_b);
        wmma_gemm<4><<<g128, 32, 0, stream>>>(qin_b, HH_, wq_b + i * HW, HH_,
                                              bq + i * HH_, nullptr, 0,
                                              q_f, HH_, q_b, HH_, HH_, 0);
        wmma_gemm<4><<<g128, 32, 0, stream>>>(seqs_b, HH_, wk_b + i * HW, HH_,
                                              bk + i * HH_, nullptr, 0,
                                              k_f, HH_, nullptr, HH_, HH_, 0);
        wmma_gemm<4><<<g128, 32, 0, stream>>>(seqs_b, HH_, wv_b + i * HW, HH_,
                                              bv + i * HH_, nullptr, 0,
                                              v_f, HH_, nullptr, HH_, HH_, 0);
        for (int n = 0; n < NHH; ++n) {
            wmma_gemm<4><<<gqt, 32, 0, stream>>>(q_b + n * HDD, HH_, tk_b + n * HDD, HH_,
                                                 nullptr, nullptr, 0,
                                                 qt_f + (size_t)n * MM * QT_LD, QT_LD,
                                                 nullptr, TS1, HDD, 0);
        }
        attn_kernel<<<gat, 32, 0, stream>>>(q_f, k_f, v_f, posK, posV, timeV,
                                            tmat, log_seqs, qt_f, ao_f);
        add_kernel<<<(nEl + 255) / 256, 256, 0, stream>>>(qin_f, ao_f, nEl);
        ln_kernel<<<MM, 32, 0, stream>>>(ao_f, fwd_g + i * HH_, fwd_b + i * HH_,
                                         seqs_f, seqs_b);
        wmma_gemm<4><<<g128, 32, 0, stream>>>(seqs_b, HH_, w1_b + i * HW, HH_,
                                              b1 + i * HH_, nullptr, 0,
                                              ff1_f, HH_, ff1_b, HH_, HH_, 1);
        wmma_gemm<4><<<g128, 32, 0, stream>>>(ff1_b, HH_, w2_b + i * HW, HH_,
                                              b2 + i * HH_, seqs_f, HH_,
                                              ao_f, HH_, nullptr, HH_, HH_, 0);
        maskmul<<<MM, HH_, 0, stream>>>(ao_f, log_seqs, seqs_f, seqs_b);
    }

    ln_kernel<<<MM, 32, 0, stream>>>(seqs_f, last_g, last_b, qin_f, nullptr);
    logits_kernel<<<MM, 32, 0, stream>>>(qin_f, item_emb, pos_seqs, neg_seqs,
                                         (float*)d_out);
}
